// HyperMambaBlock_10960756539691
// MI455X (gfx1250) — matrix operand order repaired and checked by reading, not compiled, not run on hardware
//
#include <hip/hip_runtime.h>
#include <hip/hip_bf16.h>

// ---------------------------------------------------------------------------
// HyperMambaBlock for MI455X (gfx1250), round 3:
//  - all WMMA fragments (including conv3d A) filled by b128 vector loads:
//    conv3d reads neighbor taps as row offsets into the channel-last seqbf
//  - out_proj + conv3d + bias fused into one kernel (no spec intermediate)
//  - bf16 WMMA f32-accum everywhere, 16x64 wave tiles
// ---------------------------------------------------------------------------

#define BB 2
#define CC 64
#define TT 64
#define HH 32
#define WW 32
#define DI 128
#define DS 16
#define DTR 4
#define MROWS (BB*HH*WW*TT)   // 131072
#define THW (TT*HH*WW)        // 65536

typedef __attribute__((ext_vector_type(16))) __bf16 bf16x16;
typedef __attribute__((ext_vector_type(8)))  float  f32x8;
typedef __attribute__((ext_vector_type(4)))  unsigned int uint4v;

__device__ __forceinline__ __bf16 f2bf(float f) {
  unsigned u; __builtin_memcpy(&u, &f, 4);
  unsigned r = (u + 0x7FFFu + ((u >> 16) & 1u)) >> 16;   // round-to-nearest-even
  unsigned short h = (unsigned short)r;
  __bf16 o; __builtin_memcpy(&o, &h, 2);
  return o;
}

__device__ __forceinline__ f32x8 wmma_bf16(bf16x16 a, bf16x16 b, f32x8 c) {
  return __builtin_amdgcn_wmma_f32_16x16x32_bf16(false, a, false, b, (short)0, c,
                                                 false, false);
}

// fragment = two contiguous 8-element bf16 runs, loaded as b128 each
__device__ __forceinline__ bf16x16 load_frag(const __bf16* __restrict__ p0,
                                             const __bf16* __restrict__ p1) {
  union { uint4v u[2]; bf16x16 v; } cv;
  cv.u[0] = *(const uint4v*)p0;
  cv.u[1] = *(const uint4v*)p1;
  return cv.v;
}

__device__ __forceinline__ bf16x16 zero_frag() {
  union { unsigned u[8]; bf16x16 v; } cv = {{0,0,0,0,0,0,0,0}};
  return cv.v;
}

__device__ __forceinline__ float sigmoidf_(float x) { return 1.f / (1.f + __expf(-x)); }
__device__ __forceinline__ float siluf_(float x)    { return x * sigmoidf_(x); }

// ---------------------------------------------------------------------------
// conversion kernels (one-time bf16 staging)
// ---------------------------------------------------------------------------
__global__ __launch_bounds__(256) void k_cvt(const float* __restrict__ s,
                                             __bf16* __restrict__ d, int n) {
  int i = blockIdx.x * 256 + threadIdx.x;
  if (i < n) d[i] = f2bf(s[i]);
}

// seqbf[m*64 + c] = x[b, c, t, h, w] for m = (b*1024+hw)*64 + t
// (this is simultaneously the mamba sequence AND a channel-last view of x)
__global__ __launch_bounds__(256) void k_cvt_seq(const float* __restrict__ x,
                                                 __bf16* __restrict__ seqbf) {
  long i = (long)blockIdx.x * 256 + threadIdx.x;   // m*64 + c, write-coalesced
  int c = (int)(i & 63);
  long m = i >> 6;
  int t = (int)(m & 63);
  int n = (int)(m >> 6);
  int b = n >> 10, hw = n & 1023;
  seqbf[i] = f2bf(x[((long)(b * CC + c)) * THW + t * 1024 + hw]);
}

// w3dt[tap*4096 + co*64 + ci] = conv3d_w[co, ci, tap]
__global__ __launch_bounds__(256) void k_cvt_w3dt(const float* __restrict__ w3d,
                                                  __bf16* __restrict__ wt) {
  int i = blockIdx.x * 256 + threadIdx.x;   // co*1728 + ci*27 + tap
  if (i >= 64 * 1728) return;
  int co = i / 1728; int r = i - co * 1728;
  int ci = r / 27;   int tap = r - ci * 27;
  wt[tap * 4096 + co * 64 + ci] = f2bf(w3d[i]);
}

// xpwbf padded to 48 rows (rows 36..47 zero)
__global__ __launch_bounds__(256) void k_cvt_xpw(const float* __restrict__ xpw,
                                                 __bf16* __restrict__ d) {
  int i = blockIdx.x * 256 + threadIdx.x;   // col*128 + k, col < 48
  if (i >= 48 * DI) return;
  int col = i >> 7;
  d[i] = f2bf(col < (DTR + 2 * DS) ? xpw[i] : 0.f);
}

// ---------------------------------------------------------------------------
// 1) in_proj: xz[m, 0:256] = seq @ in_proj_w^T ; grid(4, 1024), block 256
// ---------------------------------------------------------------------------
__global__ __launch_bounds__(256) void k_inproj(const __bf16* __restrict__ seqbf,
                                                const __bf16* __restrict__ wbf,
                                                float* __restrict__ xz) {
  const int lane = threadIdx.x & 31;
  const int wv   = threadIdx.x >> 5;
  const int mtile = blockIdx.y * 8 + wv;
  const int ng = blockIdx.x;             // 64-col group
  const int g  = lane >> 4;
  const int mr = lane & 15;
  const long arow = ((long)mtile * 16 + mr) * CC;

  f32x8 acc[4] = {};
#pragma unroll
  for (int k0 = 0; k0 < CC; k0 += 32) {
    bf16x16 af = load_frag(seqbf + arow + k0 + 8 * g,
                           seqbf + arow + k0 + 16 + 8 * g);
#pragma unroll
    for (int nt = 0; nt < 4; ++nt) {
      int col = ng * 64 + nt * 16 + mr;
      bf16x16 bfr = load_frag(wbf + col * CC + k0 + 8 * g,
                              wbf + col * CC + k0 + 16 + 8 * g);
      acc[nt] = wmma_bf16(af, bfr, acc[nt]);
    }
  }
#pragma unroll
  for (int nt = 0; nt < 4; ++nt)
#pragma unroll
    for (int v = 0; v < 8; ++v) {
      int mo = mtile * 16 + v + 8 * g;
      xz[(long)mo * 256 + ng * 64 + nt * 16 + mr] = acc[nt][v];
    }
}

// ---------------------------------------------------------------------------
// 2) causal depthwise conv1d + SiLU -> xi (f32 for scan) + xi_bf (GEMM A)
// ---------------------------------------------------------------------------
__global__ __launch_bounds__(256) void k_conv_silu(const float* __restrict__ xz,
                                                   const float* __restrict__ cw,
                                                   const float* __restrict__ cb,
                                                   float* __restrict__ xi,
                                                   __bf16* __restrict__ xibf) {
  long idx = (long)blockIdx.x * 256 + threadIdx.x;  // m*128 + d
  int  d = (int)(idx & (DI - 1));
  long m = idx >> 7;
  int  t = (int)(m & 63);
  float w0 = cw[d * 3 + 0], w1 = cw[d * 3 + 1], w2 = cw[d * 3 + 2];
  float a = cb[d] + w2 * xz[m * 256 + d];
  if (t >= 1) a += w1 * xz[(m - 1) * 256 + d];
  if (t >= 2) a += w0 * xz[(m - 2) * 256 + d];
  float v = siluf_(a);
  xi[idx] = v;
  xibf[idx] = f2bf(v);
}

// ---------------------------------------------------------------------------
// 3) x_proj: dbl[m, 0:36] (stride 48) ; grid(1024), block 256
// ---------------------------------------------------------------------------
__global__ __launch_bounds__(256) void k_xproj(const __bf16* __restrict__ xibf,
                                               const __bf16* __restrict__ xpwbf,
                                               float* __restrict__ dbl) {
  const int lane = threadIdx.x & 31;
  const int wv   = threadIdx.x >> 5;
  const int mtile = blockIdx.x * 8 + wv;
  const int g  = lane >> 4;
  const int mr = lane & 15;
  const long arow = ((long)mtile * 16 + mr) * DI;

  f32x8 acc[3] = {};
#pragma unroll
  for (int k0 = 0; k0 < DI; k0 += 32) {
    bf16x16 af = load_frag(xibf + arow + k0 + 8 * g,
                           xibf + arow + k0 + 16 + 8 * g);
#pragma unroll
    for (int nt = 0; nt < 3; ++nt) {
      int col = nt * 16 + mr;
      bf16x16 bfr = load_frag(xpwbf + col * DI + k0 + 8 * g,
                              xpwbf + col * DI + k0 + 16 + 8 * g);
      acc[nt] = wmma_bf16(af, bfr, acc[nt]);
    }
  }
#pragma unroll
  for (int nt = 0; nt < 3; ++nt)
#pragma unroll
    for (int v = 0; v < 8; ++v) {
      long mo = (long)mtile * 16 + v + 8 * g;
      dbl[mo * 48 + nt * 16 + mr] = acc[nt][v];
    }
}

// ---------------------------------------------------------------------------
// 4) selective scan; y written as bf16 for out_proj
// ---------------------------------------------------------------------------
__global__ __launch_bounds__(128) void k_scan(const float* __restrict__ xz,
                                              const float* __restrict__ xi,
                                              const float* __restrict__ dbl,
                                              const float* __restrict__ dtw,
                                              const float* __restrict__ dtb,
                                              const float* __restrict__ alog,
                                              const float* __restrict__ Dp,
                                              __bf16* __restrict__ ybf) {
  const int nidx = blockIdx.x;
  const int d = threadIdx.x;
  __shared__ float s[DTR + 2 * DS];

  float A[DS];
#pragma unroll
  for (int si = 0; si < DS; ++si) A[si] = -__expf(alog[d * DS + si]);
  const float q0 = dtw[d * 4 + 0], q1 = dtw[d * 4 + 1],
              q2 = dtw[d * 4 + 2], q3 = dtw[d * 4 + 3];
  const float bb = dtb[d], dpar = Dp[d];

  float h[DS];
#pragma unroll
  for (int si = 0; si < DS; ++si) h[si] = 0.f;

  for (int t = 0; t < TT; ++t) {
    const long m = (long)nidx * TT + t;
    if (threadIdx.x < DTR + 2 * DS) s[threadIdx.x] = dbl[m * 48 + threadIdx.x];
    __syncthreads();
    float dt = bb + q0 * s[0] + q1 * s[1] + q2 * s[2] + q3 * s[3];
    dt = (dt > 20.f) ? dt : log1pf(__expf(dt));
    const float xv = xi[m * DI + d];
    float yc = 0.f;
#pragma unroll
    for (int si = 0; si < DS; ++si) {
      h[si] = h[si] * __expf(dt * A[si]) + dt * s[DTR + si] * xv;
      yc += h[si] * s[DTR + DS + si];
    }
    const float zv = xz[m * 256 + DI + d];
    const float yv = (yc + xv * dpar) * siluf_(zv);
    ybf[m * DI + d] = f2bf(yv);
    __syncthreads();
  }
}

// ---------------------------------------------------------------------------
// 5) fused spec+spat: out_proj(y) + conv3d(x) + bias -> fused[b,c,t,h,w]
//    Rows m in (n,t) order. Conv taps are row offsets into channel-last seqbf:
//    m' = m + (kt-1) + 64*(kw-1) + 2048*(kh-1). grid(1024), block 256.
// ---------------------------------------------------------------------------
__global__ __launch_bounds__(256) void k_fused(const __bf16* __restrict__ ybf,
                                               const __bf16* __restrict__ opwbf,
                                               const __bf16* __restrict__ seqbf,
                                               const __bf16* __restrict__ wt,
                                               const float* __restrict__ b3d,
                                               float* __restrict__ fused) {
  const int lane = threadIdx.x & 31;
  const int wv   = threadIdx.x >> 5;
  const int mtile = blockIdx.x * 8 + wv;
  const int g  = lane >> 4;
  const int mr = lane & 15;

  const int m  = mtile * 16 + mr;          // row = (n, t)
  const int n  = m >> 6;
  const int t  = m & 63;
  const int b  = n >> 10;
  const int hw = n & 1023;
  const int h  = hw >> 5;
  const int w0 = hw & 31;

  // C/D output bases: addr = base_v[v] + col*THW
  int base_v[8];
#pragma unroll
  for (int v = 0; v < 8; ++v) {
    int mo = mtile * 16 + v + 8 * g;
    int n2 = mo >> 6, t2 = mo & 63, b2 = n2 >> 10, hw2 = n2 & 1023;
    base_v[v] = (b2 * CC * TT + t2) * 1024 + hw2;
  }

  // acc = conv3d bias
  f32x8 acc[4];
#pragma unroll
  for (int nt = 0; nt < 4; ++nt) {
    const float bias = b3d[nt * 16 + mr];
#pragma unroll
    for (int v = 0; v < 8; ++v) acc[nt][v] = bias;
  }

  // ---- out_proj: K = 128 over ybf ----
  const long arow = (long)m * DI;
#pragma unroll
  for (int k0 = 0; k0 < DI; k0 += 32) {
    bf16x16 af = load_frag(ybf + arow + k0 + 8 * g,
                           ybf + arow + k0 + 16 + 8 * g);
#pragma unroll
    for (int nt = 0; nt < 4; ++nt) {
      int col = nt * 16 + mr;
      bf16x16 bfr = load_frag(opwbf + col * DI + k0 + 8 * g,
                              opwbf + col * DI + k0 + 16 + 8 * g);
      acc[nt] = wmma_bf16(af, bfr, acc[nt]);
    }
  }

  // ---- conv3d: 27 taps x (K = 64 channels) over channel-last seqbf ----
  for (int kt = 0; kt < 3; ++kt)
    for (int kh = 0; kh < 3; ++kh)
      for (int kw = 0; kw < 3; ++kw) {
        const int tt = t + kt - 1, hh = h + kh - 1, ww = w0 + kw - 1;
        const bool valid = (unsigned)tt < (unsigned)TT &&
                           (unsigned)hh < (unsigned)HH &&
                           (unsigned)ww < (unsigned)WW;
        const int doff = (kt - 1) + 64 * (kw - 1) + 2048 * (kh - 1);
        const __bf16* arow_c = seqbf + (long)(m + doff) * CC;
        const int tap = (kt * 3 + kh) * 3 + kw;
        const __bf16* wb = wt + tap * 4096;
        __builtin_prefetch(wb + 4096, 0, 1);   // next tap's weights
#pragma unroll
        for (int kc = 0; kc < 2; ++kc) {
          const int kci0 = kc * 32;
          bf16x16 af = valid ? load_frag(arow_c + kci0 + 8 * g,
                                         arow_c + kci0 + 16 + 8 * g)
                             : zero_frag();
#pragma unroll
          for (int nt = 0; nt < 4; ++nt) {
            int col = nt * 16 + mr;
            bf16x16 bfr = load_frag(wb + col * 64 + kci0 + 8 * g,
                                    wb + col * 64 + kci0 + 16 + 8 * g);
            acc[nt] = wmma_bf16(af, bfr, acc[nt]);
          }
        }
      }

#pragma unroll
  for (int nt = 0; nt < 4; ++nt) {
    int col = nt * 16 + mr;
#pragma unroll
    for (int v = 0; v < 8; ++v)
      fused[(long)base_v[v] + (long)col * THW] = acc[nt][v];
  }
}

// ---------------------------------------------------------------------------
// 6) pool, 7) SE gate, 8) spatial gate + residual + LayerNorm
// ---------------------------------------------------------------------------
__global__ __launch_bounds__(256) void k_pool(const float* __restrict__ fused,
                                              float* __restrict__ pooled) {
  const int bc = blockIdx.x;
  __shared__ float red[256];
  float s = 0.f;
  for (int i = threadIdx.x; i < THW; i += 256) s += fused[(long)bc * THW + i];
  red[threadIdx.x] = s;
  __syncthreads();
  for (int st = 128; st > 0; st >>= 1) {
    if (threadIdx.x < st) red[threadIdx.x] += red[threadIdx.x + st];
    __syncthreads();
  }
  if (threadIdx.x == 0) pooled[bc] = red[0] * (1.f / (float)THW);
}

__global__ __launch_bounds__(64) void k_se(const float* __restrict__ pooled,
                                           const float* __restrict__ sp1w,
                                           const float* __restrict__ sp1b,
                                           const float* __restrict__ sp2w,
                                           const float* __restrict__ sp2b,
                                           float* __restrict__ swb) {
  const int b = blockIdx.x, c = threadIdx.x;
  __shared__ float hid[8];
  if (c < 8) {
    float s = sp1b[c];
    for (int k = 0; k < CC; ++k) s += pooled[b * CC + k] * sp1w[c * CC + k];
    hid[c] = fmaxf(s, 0.f);
  }
  __syncthreads();
  float s = sp2b[c];
#pragma unroll
  for (int j = 0; j < 8; ++j) s += hid[j] * sp2w[c * 8 + j];
  swb[b * CC + c] = sigmoidf_(s);
}

__global__ __launch_bounds__(256) void k_gateln(const float* __restrict__ fused,
                                                const float* __restrict__ x,
                                                const float* __restrict__ swb,
                                                const float* __restrict__ sa1w,
                                                const float* __restrict__ sa1b,
                                                const float* __restrict__ sa2w,
                                                const float* __restrict__ sa2b,
                                                const float* __restrict__ lng,
                                                const float* __restrict__ lnb,
                                                float* __restrict__ out) {
  const int v = blockIdx.x * 256 + threadIdx.x;   // (b, thw)
  const int b = v >> 16;
  const int thw = v & (THW - 1);

  float fc[CC];
#pragma unroll 8
  for (int c = 0; c < CC; ++c)
    fc[c] = fused[((long)(b * CC + c)) * THW + thw];

  float awl = sa2b[0];
#pragma unroll
  for (int j = 0; j < 8; ++j) {
    float s = sa1b[j];
#pragma unroll 8
    for (int c = 0; c < CC; ++c) s += fc[c] * sa1w[j * CC + c];
    awl += fmaxf(s, 0.f) * sa2w[j];
  }
  const float aw = sigmoidf_(awl);

  float mean = 0.f;
#pragma unroll 8
  for (int c = 0; c < CC; ++c) {
    fc[c] = fc[c] * swb[b * CC + c] * aw + x[((long)(b * CC + c)) * THW + thw];
    mean += fc[c];
  }
  mean *= (1.f / CC);
  float var = 0.f;
#pragma unroll 8
  for (int c = 0; c < CC; ++c) { float dl = fc[c] - mean; var += dl * dl; }
  var *= (1.f / CC);
  const float rstd = rsqrtf(var + 1e-5f);
#pragma unroll 8
  for (int c = 0; c < CC; ++c)
    out[((long)(b * CC + c)) * THW + thw] = (fc[c] - mean) * rstd * lng[c] + lnb[c];
}

// ---------------------------------------------------------------------------
extern "C" void kernel_launch(void* const* d_in, const int* in_sizes, int n_in,
                              void* d_out, int out_size, void* d_ws, size_t ws_size,
                              hipStream_t stream) {
  const float* x    = (const float*)d_in[0];
  const float* inw  = (const float*)d_in[1];
  const float* c1w  = (const float*)d_in[2];
  const float* c1b  = (const float*)d_in[3];
  const float* xpw  = (const float*)d_in[4];
  const float* dtw  = (const float*)d_in[5];
  const float* dtb  = (const float*)d_in[6];
  const float* alog = (const float*)d_in[7];
  const float* Dp   = (const float*)d_in[8];
  const float* opw  = (const float*)d_in[9];
  const float* w3d  = (const float*)d_in[10];
  const float* b3d  = (const float*)d_in[11];
  const float* sp1w = (const float*)d_in[12];
  const float* sp1b = (const float*)d_in[13];
  const float* sp2w = (const float*)d_in[14];
  const float* sp2b = (const float*)d_in[15];
  const float* sa1w = (const float*)d_in[16];
  const float* sa1b = (const float*)d_in[17];
  const float* sa2w = (const float*)d_in[18];
  const float* sa2b = (const float*)d_in[19];
  const float* lng  = (const float*)d_in[20];
  const float* lnb  = (const float*)d_in[21];
  float* out = (float*)d_out;

  char* wsb = (char*)d_ws;
  float*  xz     = (float*) (wsb + 0);            // M*256 f32  = 134,217,728 B
  float*  xi     = (float*) (wsb + 134217728);    // M*128 f32  =  67,108,864 B
  float*  dbl    = (float*) (wsb + 201326592);    // M*48  f32  =  25,165,824 B
  float*  fused  = (float*) (wsb + 226492416);    // BCTHW f32  =  33,554,432 B
  __bf16* seqbf  = (__bf16*)(wsb + 260046848);    // M*64  bf16 =  16,777,216 B
  __bf16* xibf   = (__bf16*)(wsb + 276824064);    // M*128 bf16 =  33,554,432 B
  __bf16* ybf    = (__bf16*)(wsb + 310378496);    // M*128 bf16 =  33,554,432 B
  __bf16* inwbf  = (__bf16*)(wsb + 343932928);    // 256*64     =      32,768 B
  __bf16* xpwbf  = (__bf16*)(wsb + 343965696);    // 48*128     =      12,288 B
  __bf16* opwbf  = (__bf16*)(wsb + 343977984);    // 64*128     =      16,384 B
  __bf16* w3dt   = (__bf16*)(wsb + 343994368);    // 27*64*64   =     221,184 B
  float*  pooled = (float*) (wsb + 344215552);    // 128 f32
  float*  swb    = (float*) (wsb + 344216064);    // 128 f32

  // bf16 staging
  k_cvt_seq <<<dim3(32768), 256, 0, stream>>>(x, seqbf);
  k_cvt     <<<dim3(64),    256, 0, stream>>>(inw, inwbf, 256 * CC);
  k_cvt_xpw <<<dim3(24),    256, 0, stream>>>(xpw, xpwbf);
  k_cvt     <<<dim3(32),    256, 0, stream>>>(opw, opwbf, CC * DI);
  k_cvt_w3dt<<<dim3(432),   256, 0, stream>>>(w3d, w3dt);

  // pipeline
  k_inproj   <<<dim3(4, 1024), 256, 0, stream>>>(seqbf, inwbf, xz);
  k_conv_silu<<<dim3(65536),   256, 0, stream>>>(xz, c1w, c1b, xi, xibf);
  k_xproj    <<<dim3(1024),    256, 0, stream>>>(xibf, xpwbf, dbl);
  k_scan     <<<dim3(2048),    128, 0, stream>>>(xz, xi, dbl, dtw, dtb, alog, Dp, ybf);
  k_fused    <<<dim3(1024),    256, 0, stream>>>(ybf, opwbf, seqbf, w3dt, b3d, fused);
  k_pool     <<<dim3(128),     256, 0, stream>>>(fused, pooled);
  k_se       <<<dim3(2),        64, 0, stream>>>(pooled, sp1w, sp1b, sp2w, sp2b, swb);
  k_gateln   <<<dim3(512),     256, 0, stream>>>(fused, x, swb, sa1w, sa1b, sa2w,
                                                 sa2b, lng, lnb, out);
}
